// EfficientSelfAtten_1726576854494
// MI455X (gfx1250) — matrix operand order repaired and checked by reading, not compiled, not run on hardware
//
#include <hip/hip_runtime.h>
#include <hip/hip_bf16.h>
#include <cstdint>

typedef __attribute__((ext_vector_type(16))) _Float16 v16h;
typedef __attribute__((ext_vector_type(8)))  float    v8f;
typedef __attribute__((ext_vector_type(4)))  int      v4i;

union HVec { v16h h; unsigned int u[8]; };

#define WMMA_F16(A, B, Cacc) \
  __builtin_amdgcn_wmma_f32_16x16x32_f16(false, (A), false, (B), (short)0, (Cacc), false, false)

// ---- gfx1250 async global->LDS copy (ASYNCcnt path), with sync fallback ----
#if defined(__has_builtin)
#if __has_builtin(__builtin_amdgcn_global_load_async_to_lds_b128)
#define HAVE_ASYNC_LDS 1
#endif
#endif

#ifndef HAVE_ASYNC_LDS
#define HAVE_ASYNC_LDS 0
#endif

#define AS1 __attribute__((address_space(1)))
#define AS3 __attribute__((address_space(3)))

#if HAVE_ASYNC_LDS
__device__ __forceinline__ void async_b128(const void* g, void* l) {
  // Param type per hipcc diagnostic: v4i in AS(1) ("__device__") for the
  // global source; LDS side cast to AS(3). Flat==global addresses, so the
  // AS1 pointer is materialized via inttoptr.
  __builtin_amdgcn_global_load_async_to_lds_b128(
      (AS1 v4i*)(uintptr_t)g, (AS3 v4i*)l, 0, 0);
}
__device__ __forceinline__ void wait_async0() {
#if __has_builtin(__builtin_amdgcn_s_wait_asynccnt)
  __builtin_amdgcn_s_wait_asynccnt(0);
#else
  asm volatile("s_wait_asynccnt 0x0" ::: "memory");
#endif
}
#endif

// A operand (16x32 f16, M x K): lane l holds row M = l&15.
// VGPR j holds packed K pair at K = 16*(j>>2) + 8*(l>>4) + 2*(j&3).
__device__ __forceinline__ v16h load_a16(const _Float16* rowp, int lane) {
  HVec a;
  const unsigned int* p = (const unsigned int*)rowp;
  const int hb = (lane >> 4) ? 4 : 0;
#pragma unroll
  for (int j = 0; j < 8; ++j)
    a.u[j] = p[((j >> 2) << 3) + hb + (j & 3)];
  return a.h;
}

// B operand (32x16 f16, K x N): lane l holds column N = l&15.
// VGPR j holds packed K pair at K = 16*(j>>2) + 8*(l>>4) + 2*(j&3) ([n][k] LDS layout).
__device__ __forceinline__ v16h load_b16(const _Float16* colp, int lane) {
  HVec b;
  const unsigned int* p = (const unsigned int*)colp;
  const int off = (lane >> 4) ? 4 : 0;
#pragma unroll
  for (int j = 0; j < 8; ++j) b.u[j] = p[((j >> 2) << 3) + off + (j & 3)];
  return b.h;
}

// ---------------------------------------------------------------------------
// Weight conversion: fp32 -> f16; sr_w (OIHW, O=256, I*Kh*Kw=4096) transposed
// to [k][o] so GEMM B-tiles read contiguously.
// ---------------------------------------------------------------------------
__global__ void __launch_bounds__(256) convert_weights(
    const float* __restrict__ qw, const float* __restrict__ kvw,
    const float* __restrict__ pw, const float* __restrict__ srw,
    _Float16* __restrict__ qh, _Float16* __restrict__ kvh,
    _Float16* __restrict__ ph, _Float16* __restrict__ srt) {
  unsigned int i = blockIdx.x * 256u + threadIdx.x;
  if (i < 65536u)  qh[i]  = (_Float16)qw[i];
  if (i < 131072u) kvh[i] = (_Float16)kvw[i];
  if (i < 65536u)  ph[i]  = (_Float16)pw[i];
  if (i < 1048576u) {
    unsigned int o = i & 255u, k = i >> 8;
    srt[i] = (_Float16)srw[(size_t)o * 4096 + k];
  }
}

// ---------------------------------------------------------------------------
// Q projection: (32768 x 256) fp32 x (256 x 256) f16 -> f16, +bias.
// WG = 128 threads (4 waves), 64x64 output tile, K step 32.
// ---------------------------------------------------------------------------
__global__ void __launch_bounds__(128) gemm_qproj(
    const float* __restrict__ x, const _Float16* __restrict__ wh,
    const float* __restrict__ bias, _Float16* __restrict__ out) {
  __shared__ __align__(16) _Float16 As[64 * 32];   // [m][k]
  __shared__ __align__(16) _Float16 Bs[64 * 32];   // [n][k]
  const int t = threadIdx.x, lane = t & 31, wave = t >> 5;
  const int rowBlk = blockIdx.y * 64, colBlk = blockIdx.x * 64;
  v8f acc[4] = {};
  for (int k0 = 0; k0 < 256; k0 += 32) {
    if (k0 + 32 < 256)
      __builtin_prefetch(&x[(size_t)(rowBlk + (t >> 1)) * 256 + k0 + 32], 0, 3);
#pragma unroll
    for (int i = 0; i < 16; ++i) {
      int idx = i * 128 + t, m = idx >> 5, k = idx & 31;
      As[idx] = (_Float16)x[(size_t)(rowBlk + m) * 256 + k0 + k];
    }
#pragma unroll
    for (int i = 0; i < 16; ++i) {
      int idx = i * 128 + t, k = idx >> 6, n = idx & 63;
      Bs[n * 32 + k] = wh[(size_t)(k0 + k) * 256 + colBlk + n];
    }
    __syncthreads();
    v16h av = load_a16(As + (wave * 16 + (lane & 15)) * 32, lane);
#pragma unroll
    for (int nt = 0; nt < 4; ++nt) {
      v16h bv = load_b16(Bs + (nt * 16 + (lane & 15)) * 32, lane);
      acc[nt] = WMMA_F16(av, bv, acc[nt]);
    }
    __syncthreads();
  }
  const int nloc = lane & 15, gm = (lane >> 4) * 8;
#pragma unroll
  for (int nt = 0; nt < 4; ++nt) {
    int col = colBlk + nt * 16 + nloc;
    float bv = bias[col];
#pragma unroll
    for (int v = 0; v < 8; ++v) {
      int row = rowBlk + wave * 16 + gm + v;
      out[(size_t)row * 256 + col] = (_Float16)(acc[nt][v] + bv);
    }
  }
}

// ---------------------------------------------------------------------------
// Spatial-reduction conv as gather-GEMM: rows = (b, oy, ox) = 2048,
// K = (ci, ky, kx) = 4096, N = 256 out channels. fp32 output for layernorm.
// ---------------------------------------------------------------------------
__global__ void __launch_bounds__(128) gemm_conv(
    const float* __restrict__ x, const _Float16* __restrict__ wh,
    const float* __restrict__ bias, float* __restrict__ out) {
  __shared__ __align__(16) _Float16 As[64 * 32];
  __shared__ __align__(16) _Float16 Bs[64 * 32];
  const int t = threadIdx.x, lane = t & 31, wave = t >> 5;
  const int rowBlk = blockIdx.y * 64, colBlk = blockIdx.x * 64;
  v8f acc[4] = {};
  for (int k0 = 0; k0 < 4096; k0 += 32) {
#pragma unroll
    for (int i = 0; i < 16; ++i) {
      int idx = i * 128 + t, m = idx >> 5, k = idx & 31;
      int r = rowBlk + m;
      int b = r >> 8, rr = r & 255, oy = rr >> 4, ox = rr & 15;
      int kg = k0 + k;
      int ci = kg >> 4, rem = kg & 15, ky = rem >> 2, kx = rem & 3;
      size_t src = ((size_t)b * 4096 + (oy * 4 + ky) * 64 + (ox * 4 + kx)) * 256 + ci;
      As[idx] = (_Float16)x[src];
    }
#pragma unroll
    for (int i = 0; i < 16; ++i) {
      int idx = i * 128 + t, k = idx >> 6, n = idx & 63;
      Bs[n * 32 + k] = wh[(size_t)(k0 + k) * 256 + colBlk + n];
    }
    __syncthreads();
    v16h av = load_a16(As + (wave * 16 + (lane & 15)) * 32, lane);
#pragma unroll
    for (int nt = 0; nt < 4; ++nt) {
      v16h bv = load_b16(Bs + (nt * 16 + (lane & 15)) * 32, lane);
      acc[nt] = WMMA_F16(av, bv, acc[nt]);
    }
    __syncthreads();
  }
  const int nloc = lane & 15, gm = (lane >> 4) * 8;
#pragma unroll
  for (int nt = 0; nt < 4; ++nt) {
    int col = colBlk + nt * 16 + nloc;
    float bv = bias[col];
#pragma unroll
    for (int v = 0; v < 8; ++v) {
      int row = rowBlk + wave * 16 + gm + v;
      out[(size_t)row * 256 + col] = acc[nt][v] + bv;
    }
  }
}

// ---------------------------------------------------------------------------
// LayerNorm over 256 channels per reduced token (2048 rows), fp32 in, f16 out.
// ---------------------------------------------------------------------------
__global__ void __launch_bounds__(256) ln_kernel(
    const float* __restrict__ xr, const float* __restrict__ g,
    const float* __restrict__ bta, _Float16* __restrict__ out) {
  const int t = threadIdx.x, lane = t & 31, wave = t >> 5;
  const int row = blockIdx.x * 8 + wave;
  const float* p = xr + (size_t)row * 256;
  float vals[8];
  float s = 0.f;
#pragma unroll
  for (int i = 0; i < 8; ++i) { vals[i] = p[i * 32 + lane]; s += vals[i]; }
#pragma unroll
  for (int m = 1; m < 32; m <<= 1) s += __shfl_xor(s, m, 32);
  const float mu = s * (1.0f / 256.0f);
  float vs = 0.f;
#pragma unroll
  for (int i = 0; i < 8; ++i) { float d = vals[i] - mu; vs += d * d; }
#pragma unroll
  for (int m = 1; m < 32; m <<= 1) vs += __shfl_xor(vs, m, 32);
  const float r = rsqrtf(vs * (1.0f / 256.0f) + 1e-5f);
#pragma unroll
  for (int i = 0; i < 8; ++i) {
    int c = i * 32 + lane;
    out[(size_t)row * 256 + c] = (_Float16)((vals[i] - mu) * r * g[c] + bta[c]);
  }
}

// ---------------------------------------------------------------------------
// KV projection: (2048 x 256) f16 x (256 x 512) f16, scatter into per-(b,h)
// K/V buffers [bh][m][d] f16. A-tile fill uses async global->LDS when present.
// ---------------------------------------------------------------------------
__global__ void __launch_bounds__(128) gemm_kv(
    const _Float16* __restrict__ a, const _Float16* __restrict__ wh,
    const float* __restrict__ bias, _Float16* __restrict__ Kh,
    _Float16* __restrict__ Vh) {
  __shared__ __align__(16) _Float16 As[64 * 32];
  __shared__ __align__(16) _Float16 Bs[64 * 32];
  const int t = threadIdx.x, lane = t & 31, wave = t >> 5;
  const int rowBlk = blockIdx.y * 64, colBlk = blockIdx.x * 64;
  v8f acc[4] = {};
  const unsigned int* s32 = (const unsigned int*)a;
  unsigned int* a32 = (unsigned int*)As;
  for (int k0 = 0; k0 < 256; k0 += 32) {
#if HAVE_ASYNC_LDS
#pragma unroll
    for (int i = 0; i < 2; ++i) {
      int w = (i * 128 + t) * 4;            // u32 index, 16B aligned
      int m = w >> 4, kw = w & 15;
      async_b128(s32 + (size_t)(rowBlk + m) * 128 + (k0 >> 1) + kw, a32 + w);
    }
#else
#pragma unroll
    for (int i = 0; i < 8; ++i) {
      int w = i * 128 + t, m = w >> 4, kw = w & 15;
      a32[w] = s32[(size_t)(rowBlk + m) * 128 + (k0 >> 1) + kw];
    }
#endif
#pragma unroll
    for (int i = 0; i < 16; ++i) {
      int idx = i * 128 + t, k = idx >> 6, n = idx & 63;
      Bs[n * 32 + k] = wh[(size_t)(k0 + k) * 512 + colBlk + n];
    }
#if HAVE_ASYNC_LDS
    wait_async0();
#endif
    __syncthreads();
    v16h av = load_a16(As + (wave * 16 + (lane & 15)) * 32, lane);
#pragma unroll
    for (int nt = 0; nt < 4; ++nt) {
      v16h bv = load_b16(Bs + (nt * 16 + (lane & 15)) * 32, lane);
      acc[nt] = WMMA_F16(av, bv, acc[nt]);
    }
    __syncthreads();
  }
  const int nloc = lane & 15, gm = (lane >> 4) * 8;
#pragma unroll
  for (int nt = 0; nt < 4; ++nt) {
    int col = colBlk + nt * 16 + nloc;           // 0..511
    int sel = col >> 8, hh = (col >> 5) & 7, d = col & 31;
    float bv = bias[col];
    _Float16* dst = sel ? Vh : Kh;
#pragma unroll
    for (int v = 0; v < 8; ++v) {
      int row = rowBlk + wave * 16 + gm + v;     // 0..2047
      int b = row >> 8, mm = row & 255;
      dst[((size_t)(b * 8 + hh) * 256 + mm) * 32 + d] = (_Float16)(acc[nt][v] + bv);
    }
  }
}

// ---------------------------------------------------------------------------
// Fused attention per (b, h, 64-query block). K (256x32) async-staged to LDS,
// V^T (32x256) sync-transposed into LDS; online (flash) softmax; P converted
// C-layout -> A-layout through a 1KB/wave LDS staging buffer.
// ---------------------------------------------------------------------------
__global__ void __launch_bounds__(128) attn_kernel(
    const _Float16* __restrict__ Qh, const _Float16* __restrict__ Kh,
    const _Float16* __restrict__ Vh, _Float16* __restrict__ Oh) {
  __shared__ __align__(16) _Float16 Kl[256 * 32];        // [m][d]
  __shared__ __align__(16) _Float16 Vt[32 * 256];        // [d][m]
  __shared__ __align__(16) _Float16 Pst[4][16 * 32];     // per-wave P staging
  const int t = threadIdx.x, lane = t & 31, wave = t >> 5;
  const int blk = blockIdx.x;              // 0..4095
  const int bh = blk >> 6;                 // 0..63
  const int qblk = (blk & 63) * 64;
  const int b = bh >> 3, h = bh & 7;

#if HAVE_ASYNC_LDS
  { // K tile: 16KB contiguous copy via async LDS fill (ASYNCcnt path)
    const char* ks = (const char*)(Kh + (size_t)bh * 256 * 32);
    char* kd = (char*)Kl;
#pragma unroll
    for (int i = 0; i < 8; ++i) {
      int off = (i * 128 + t) * 16;
      async_b128(ks + off, kd + off);
    }
  }
#else
  {
    const unsigned int* ks = (const unsigned int*)(Kh + (size_t)bh * 256 * 32);
    unsigned int* kd = (unsigned int*)Kl;
#pragma unroll
    for (int i = 0; i < 32; ++i) kd[i * 128 + t] = ks[i * 128 + t];
  }
#endif
  { // V load transposed to [d][m] so token pairs pack into u32 B operands
    const _Float16* vs = Vh + (size_t)bh * 256 * 32;
#pragma unroll
    for (int i = 0; i < 64; ++i) {
      int idx = i * 128 + t, m = idx >> 5, d = idx & 31;
      Vt[d * 256 + m] = vs[idx];
    }
  }
#if HAVE_ASYNC_LDS
  wait_async0();
#endif
  __syncthreads();

  // Q A-operand: per-lane global u32 gathers (row-major f16 Q).
  HVec qa;
  {
    const int qrow = qblk + wave * 16 + (lane & 15);
    const size_t qbase = ((size_t)b * 4096 + qrow) * 256 + h * 32;
    const int hb = (lane >> 4) ? 8 : 0;
#pragma unroll
    for (int j = 0; j < 8; ++j) {
      int kb = ((j >> 2) << 4) + hb + ((j & 3) << 1);
      qa.u[j] = *(const unsigned int*)(Qh + qbase + kb);
    }
  }

  const float scale = 0.17677669529663687f;  // 32^-0.5
  float mx[8], sm[8];
#pragma unroll
  for (int v = 0; v < 8; ++v) { mx[v] = -1e30f; sm[v] = 0.f; }
  v8f dacc[2] = {};
  _Float16* stag = &Pst[wave][0];

  for (int mt = 0; mt < 16; mt += 2) {
    v8f s0 = {}, s1 = {};
    {
      v16h kb0 = load_b16(Kl + (mt * 16 + (lane & 15)) * 32, lane);
      v16h kb1 = load_b16(Kl + ((mt + 1) * 16 + (lane & 15)) * 32, lane);
      s0 = WMMA_F16(qa.h, kb0, s0);
      s1 = WMMA_F16(qa.h, kb1, s1);
    }
    float p0[8], p1[8], fs[8];
#pragma unroll
    for (int v = 0; v < 8; ++v) {
      float t0 = s0[v] * scale, t1 = s1[v] * scale;
      float tm = fmaxf(t0, t1);
      tm = fmaxf(tm, __shfl_xor(tm, 1, 32));
      tm = fmaxf(tm, __shfl_xor(tm, 2, 32));
      tm = fmaxf(tm, __shfl_xor(tm, 4, 32));
      tm = fmaxf(tm, __shfl_xor(tm, 8, 32));
      float nm = fmaxf(mx[v], tm);
      float f = __expf(mx[v] - nm);
      p0[v] = __expf(t0 - nm);
      p1[v] = __expf(t1 - nm);
      float ps = p0[v] + p1[v];
      ps += __shfl_xor(ps, 1, 32);
      ps += __shfl_xor(ps, 2, 32);
      ps += __shfl_xor(ps, 4, 32);
      ps += __shfl_xor(ps, 8, 32);
      sm[v] = sm[v] * f + ps;
      mx[v] = nm;
      fs[v] = f;
    }
#pragma unroll
    for (int v = 0; v < 8; ++v) { dacc[0][v] *= fs[v]; dacc[1][v] *= fs[v]; }
    __syncthreads();  // staging reused: previous reads complete
    {
      const int nloc = lane & 15, gm = (lane >> 4) * 8;
#pragma unroll
      for (int v = 0; v < 8; ++v) {
        stag[(gm + v) * 32 + nloc]      = (_Float16)p0[v];
        stag[(gm + v) * 32 + 16 + nloc] = (_Float16)p1[v];
      }
    }
    __syncthreads();
    v16h pa = load_a16(stag + (lane & 15) * 32, lane);
#pragma unroll
    for (int dt = 0; dt < 2; ++dt) {
      HVec bv;
      const int d = dt * 16 + (lane & 15);
      const unsigned int* vp = (const unsigned int*)(Vt + d * 256 + mt * 16);
      const int off = (lane >> 4) ? 4 : 0;
#pragma unroll
      for (int j = 0; j < 8; ++j) bv.u[j] = vp[((j >> 2) << 3) + off + (j & 3)];
      dacc[dt] = WMMA_F16(pa, bv.h, dacc[dt]);
    }
  }

  {
    const int nloc = lane & 15, gm = (lane >> 4) * 8;
#pragma unroll
    for (int v = 0; v < 8; ++v) {
      float rinv = 1.0f / sm[v];
      int row = qblk + wave * 16 + gm + v;
      size_t o = ((size_t)b * 4096 + row) * 256 + h * 32;
      Oh[o + nloc]      = (_Float16)(dacc[0][v] * rinv);
      Oh[o + 16 + nloc] = (_Float16)(dacc[1][v] * rinv);
    }
  }
}

// ---------------------------------------------------------------------------
// Output projection: (32768 x 256) f16 x (256 x 256) f16 -> fp32 + bias.
// ---------------------------------------------------------------------------
__global__ void __launch_bounds__(128) gemm_proj(
    const _Float16* __restrict__ a, const _Float16* __restrict__ wh,
    const float* __restrict__ bias, float* __restrict__ out) {
  __shared__ __align__(16) _Float16 As[64 * 32];
  __shared__ __align__(16) _Float16 Bs[64 * 32];
  const int t = threadIdx.x, lane = t & 31, wave = t >> 5;
  const int rowBlk = blockIdx.y * 64, colBlk = blockIdx.x * 64;
  v8f acc[4] = {};
  const unsigned int* s32 = (const unsigned int*)a;
  unsigned int* a32 = (unsigned int*)As;
  for (int k0 = 0; k0 < 256; k0 += 32) {
#if HAVE_ASYNC_LDS
#pragma unroll
    for (int i = 0; i < 2; ++i) {
      int w = (i * 128 + t) * 4;
      int m = w >> 4, kw = w & 15;
      async_b128(s32 + (size_t)(rowBlk + m) * 128 + (k0 >> 1) + kw, a32 + w);
    }
#else
#pragma unroll
    for (int i = 0; i < 8; ++i) {
      int w = i * 128 + t, m = w >> 4, kw = w & 15;
      a32[w] = s32[(size_t)(rowBlk + m) * 128 + (k0 >> 1) + kw];
    }
#endif
#pragma unroll
    for (int i = 0; i < 16; ++i) {
      int idx = i * 128 + t, k = idx >> 6, n = idx & 63;
      Bs[n * 32 + k] = wh[(size_t)(k0 + k) * 256 + colBlk + n];
    }
#if HAVE_ASYNC_LDS
    wait_async0();
#endif
    __syncthreads();
    v16h av = load_a16(As + (wave * 16 + (lane & 15)) * 32, lane);
#pragma unroll
    for (int nt = 0; nt < 4; ++nt) {
      v16h bv = load_b16(Bs + (nt * 16 + (lane & 15)) * 32, lane);
      acc[nt] = WMMA_F16(av, bv, acc[nt]);
    }
    __syncthreads();
  }
  const int nloc = lane & 15, gm = (lane >> 4) * 8;
#pragma unroll
  for (int nt = 0; nt < 4; ++nt) {
    int col = colBlk + nt * 16 + nloc;
    float bv = bias[col];
#pragma unroll
    for (int v = 0; v < 8; ++v) {
      int row = rowBlk + wave * 16 + gm + v;
      out[(size_t)row * 256 + col] = acc[nt][v] + bv;
    }
  }
}

// ---------------------------------------------------------------------------

extern "C" void kernel_launch(void* const* d_in, const int* in_sizes, int n_in,
                              void* d_out, int out_size, void* d_ws, size_t ws_size,
                              hipStream_t stream) {
  (void)in_sizes; (void)n_in; (void)out_size; (void)ws_size;
  const float* x      = (const float*)d_in[0];
  // d_in[1]=H, d_in[2]=W (compile-time constants 64 here)
  const float* q_w    = (const float*)d_in[3];
  const float* q_b    = (const float*)d_in[4];
  const float* kv_w   = (const float*)d_in[5];
  const float* kv_b   = (const float*)d_in[6];
  const float* proj_w = (const float*)d_in[7];
  const float* proj_b = (const float*)d_in[8];
  const float* sr_w   = (const float*)d_in[9];
  const float* sr_b   = (const float*)d_in[10];
  const float* norm_g = (const float*)d_in[11];
  const float* norm_b = (const float*)d_in[12];
  float* out = (float*)d_out;

  char* ws = (char*)d_ws;
  size_t off = 0;
  auto alloc = [&](size_t bytes) -> char* {
    char* p = ws + off;
    off += (bytes + 255) & ~(size_t)255;
    return p;
  };
  _Float16* Qh    = (_Float16*)alloc((size_t)8 * 4096 * 256 * 2);  // 16 MB
  _Float16* Oh    = (_Float16*)alloc((size_t)8 * 4096 * 256 * 2);  // 16 MB
  _Float16* wq_h  = (_Float16*)alloc((size_t)65536 * 2);
  _Float16* wkv_h = (_Float16*)alloc((size_t)131072 * 2);
  _Float16* wp_h  = (_Float16*)alloc((size_t)65536 * 2);
  _Float16* wsr_t = (_Float16*)alloc((size_t)1048576 * 2);
  float*    xr    = (float*)alloc((size_t)2048 * 256 * 4);
  _Float16* xrh   = (_Float16*)alloc((size_t)2048 * 256 * 2);
  _Float16* Kh    = (_Float16*)alloc((size_t)64 * 256 * 32 * 2);
  _Float16* Vh    = (_Float16*)alloc((size_t)64 * 256 * 32 * 2);

  convert_weights<<<4096, 256, 0, stream>>>(q_w, kv_w, proj_w, sr_w,
                                            wq_h, wkv_h, wp_h, wsr_t);
  gemm_qproj<<<dim3(4, 512), 128, 0, stream>>>(x, wq_h, q_b, Qh);
  gemm_conv<<<dim3(4, 32), 128, 0, stream>>>(x, wsr_t, sr_b, xr);
  ln_kernel<<<256, 256, 0, stream>>>(xr, norm_g, norm_b, xrh);
  gemm_kv<<<dim3(8, 32), 128, 0, stream>>>(xrh, wkv_h, kv_b, Kh, Vh);
  attn_kernel<<<4096, 128, 0, stream>>>(Qh, Kh, Vh, Oh);
  gemm_proj<<<dim3(4, 512), 128, 0, stream>>>(Oh, wp_h, proj_b, out);
}
